// LSTM_5188320494082
// MI455X (gfx1250) — compile-verified, matrix operand order checked by reading, not run on hardware
//
#include <hip/hip_runtime.h>
#include <hip/hip_bf16.h>

// ---------------------------------------------------------------------------
// LSTM on MI455X (gfx1250): bf16 WMMA (v_wmma_f32_16x16x32_bf16) with fp32
// accumulate. Phase 1: big GEMM xW = x@Wx + b (register double-buffered,
// 32x64 per wave, unroll-2 so buffer rotation is register renaming, not
// v_mov copies). Phase 2: 512 serial step kernels fusing h_prev@Wh with the
// gate math; xW_t epilogue data is staged via async global->LDS copies
// overlapped with the K loop.
// ---------------------------------------------------------------------------

typedef __attribute__((ext_vector_type(16))) __bf16 v16bf;
typedef __attribute__((ext_vector_type(8)))  __bf16 v8bf;
typedef __attribute__((ext_vector_type(8)))  float  v8f;

// Types for the async global->LDS builtin (from the compiler diagnostic:
// param0 = 'int __vector(4) __device__ *'  i.e. vector_size(16) int, as(1)).
typedef int v4i_vs __attribute__((vector_size(16)));
typedef v4i_vs __attribute__((address_space(1)))* g_v4i_p;
typedef v4i_vs __attribute__((address_space(3)))* l_v4i_p;

#define NB   64      // batch
#define TT   512     // timesteps
#define DD   1024    // input dim
#define HH   1024    // hidden dim
#define FH   4096    // 4*H

#if __has_builtin(__builtin_amdgcn_global_load_async_to_lds_b128)
#define HAVE_ASYNC_LDS 1
#else
#define HAVE_ASYNC_LDS 0
#endif

__device__ __forceinline__ v16bf load_frag(const __bf16* p0, const __bf16* p1) {
    v8bf lo = *(const v8bf*)p0;
    v8bf hi = *(const v8bf*)p1;
    return __builtin_shufflevector(lo, hi, 0,1,2,3,4,5,6,7,8,9,10,11,12,13,14,15);
}

__device__ __forceinline__ float sigmoidf_(float x) {
    return 1.0f / (1.0f + __expf(-x));
}

// ---------------- elementwise fp32 -> bf16 convert -------------------------
__global__ __launch_bounds__(256)
void cvt_f32_bf16(const float* __restrict__ src, __bf16* __restrict__ dst, int n) {
    int i = (blockIdx.x * 256 + threadIdx.x) * 4;
    if (i + 3 < n) {
        float4 v = *(const float4*)(src + i);
        dst[i + 0] = (__bf16)v.x;
        dst[i + 1] = (__bf16)v.y;
        dst[i + 2] = (__bf16)v.z;
        dst[i + 3] = (__bf16)v.w;
    } else {
        for (int k = i; k < n; ++k) dst[k] = (__bf16)src[k];
    }
}

// ---------------- zero fp32 buffer -----------------------------------------
__global__ __launch_bounds__(256)
void zero_f32(float* __restrict__ dst, int n) {
    int i = blockIdx.x * 256 + threadIdx.x;
    if (i < n) dst[i] = 0.0f;
}

// ---------------- LDS-tiled transpose + convert: W[K][C] -> WT[C][K] -------
__global__ __launch_bounds__(256)
void transpose_cvt(const float* __restrict__ W, __bf16* __restrict__ WT,
                   int rows, int cols) {
    __shared__ __bf16 tile[32][33];
    int cb = blockIdx.x * 32;
    int rb = blockIdx.y * 32;
    int tx = threadIdx.x & 31;
    int ty = threadIdx.x >> 5;
    #pragma unroll
    for (int r = 0; r < 32; r += 8)
        tile[r + ty][tx] = (__bf16)W[(size_t)(rb + r + ty) * cols + cb + tx];
    __syncthreads();
    #pragma unroll
    for (int r = 0; r < 32; r += 8)
        WT[(size_t)(cb + r + ty) * rows + rb + tx] = tile[tx][r + ty];
}

// ---------------- Phase 1 GEMM: xW[t][n][col] = x@Wx + b -------------------
// Each wave: 32(M) x 64(N) tile = 8 accumulators, double-buffered fragments.
// Block = 4 waves = 128(M) x 64(N). grid = (256, 64).
__global__ __launch_bounds__(128)
void gemm_xw(const __bf16* __restrict__ xb, const __bf16* __restrict__ WxT,
             const float* __restrict__ bias, float* __restrict__ xW) {
    int lane = threadIdx.x & 31;
    int wave = threadIdx.x >> 5;
    int m0  = blockIdx.x * 128 + wave * 32;
    int nb0 = blockIdx.y * 64;
    int lr  = lane & 15;
    int kh  = (lane >> 4) * 8;

    const __bf16* Ar0 = xb + (size_t)(m0 + lr) * DD + kh;
    const __bf16* Ar1 = xb + (size_t)(m0 + 16 + lr) * DD + kh;
    const __bf16* Br[4];
    #pragma unroll
    for (int s = 0; s < 4; ++s)
        Br[s] = WxT + (size_t)(nb0 + s * 16 + lr) * DD + kh;

    v8f acc[2][4] = {};

    // prime current fragments (chunk 0)
    v16bf a0 = load_frag(Ar0, Ar0 + 16);
    v16bf a1 = load_frag(Ar1, Ar1 + 16);
    v16bf bc[4];
    #pragma unroll
    for (int s = 0; s < 4; ++s) bc[s] = load_frag(Br[s], Br[s] + 16);

    // unroll 2: ping-pong buffers become SSA renames, not v_mov copies
    #pragma unroll 2
    for (int k0 = 0; k0 < DD - 32; k0 += 32) {
        const int kn = k0 + 32;
        v16bf a0n = load_frag(Ar0 + kn, Ar0 + kn + 16);
        v16bf a1n = load_frag(Ar1 + kn, Ar1 + kn + 16);
        v16bf bn[4];
        #pragma unroll
        for (int s = 0; s < 4; ++s) bn[s] = load_frag(Br[s] + kn, Br[s] + kn + 16);
        #pragma unroll
        for (int s = 0; s < 4; ++s) {
            acc[0][s] = __builtin_amdgcn_wmma_f32_16x16x32_bf16(
                false, a0, false, bc[s], (short)0, acc[0][s], false, false);
            acc[1][s] = __builtin_amdgcn_wmma_f32_16x16x32_bf16(
                false, a1, false, bc[s], (short)0, acc[1][s], false, false);
        }
        a0 = a0n; a1 = a1n;
        #pragma unroll
        for (int s = 0; s < 4; ++s) bc[s] = bn[s];
    }
    #pragma unroll
    for (int s = 0; s < 4; ++s) {
        acc[0][s] = __builtin_amdgcn_wmma_f32_16x16x32_bf16(
            false, a0, false, bc[s], (short)0, acc[0][s], false, false);
        acc[1][s] = __builtin_amdgcn_wmma_f32_16x16x32_bf16(
            false, a1, false, bc[s], (short)0, acc[1][s], false, false);
    }

    // C/D layout: VGPR i -> row tile_m0 + i + 8*(lane/16), col = base + lane%16
    #pragma unroll
    for (int mu = 0; mu < 2; ++mu) {
        #pragma unroll
        for (int s = 0; s < 4; ++s) {
            int col = nb0 + s * 16 + lr;
            float bv = bias[col];
            #pragma unroll
            for (int i = 0; i < 8; ++i) {
                int m = m0 + mu * 16 + i + ((lane >> 4) << 3);
                int t = m & (TT - 1);
                int n = m >> 9;           // m / TT
                __builtin_nontemporal_store(acc[mu][s][i] + bv,
                    &xW[((size_t)t * NB + n) * FH + col]);
            }
        }
    }
}

// ---------------- Phase 2: one recurrent step ------------------------------
// grid.x = H/16 = 64 blocks; 4 waves/block cover all 64 batch rows.
// Wave accumulators = gates i,f,o,g for the same 16x16 patch. xW_t tile
// (64 rows x 4x16 gate cols = 16KB) is async-copied to LDS at kernel start
// and only waited on right before the pointwise epilogue.
__global__ __launch_bounds__(128)
void lstm_step(const float* __restrict__ xW_t,   // [64, 4096] this timestep
               const __bf16* __restrict__ h_prev,// [64, 1024]
               const __bf16* __restrict__ WhT,   // [4096, 1024]
               float* __restrict__ c,            // [64, 1024]
               __bf16* __restrict__ h_next,      // [64, 1024]
               float* __restrict__ out,          // [N, T, H]
               int t) {
    __shared__ float xw_tile[NB * 64];           // [row][g*16 + cwithin]
    int lane = threadIdx.x & 31;
    int wave = threadIdx.x >> 5;
    int m0 = wave * 16;                  // batch-row tile
    int cb = blockIdx.x * 16;            // hidden-col tile
    int lr = lane & 15;
    int kh = (lane >> 4) * 8;

    // Stage xW_t[row][g*HH + cb .. +16) for all 64 rows, 4 gates into LDS.
    // 1024 x 16B chunks, 8 per thread.
    {
        int tid = threadIdx.x;
        #pragma unroll
        for (int q = 0; q < 8; ++q) {
            int idx = tid + q * 128;          // 0..1023
            int row = idx >> 4;
            int g   = (idx >> 2) & 3;
            int sub = idx & 3;
            const float* gsrc = xW_t + (size_t)row * FH + g * HH + cb + sub * 4;
            float* ldst = &xw_tile[row * 64 + g * 16 + sub * 4];
#if HAVE_ASYNC_LDS
            __builtin_amdgcn_global_load_async_to_lds_b128(
                (g_v4i_p)gsrc, (l_v4i_p)ldst, 0, 0);
#else
            *(float4*)ldst = *(const float4*)gsrc;
#endif
        }
    }

    const __bf16* Arow = h_prev + (size_t)(m0 + lr) * HH + kh;
    const __bf16* Br[4];
    #pragma unroll
    for (int g = 0; g < 4; ++g)
        Br[g] = WhT + (size_t)(g * HH + cb + lr) * HH + kh;

    v8f acc[4] = {};
    v16bf a = load_frag(Arow, Arow + 16);
    v16bf bc[4];
    #pragma unroll
    for (int g = 0; g < 4; ++g) bc[g] = load_frag(Br[g], Br[g] + 16);

    #pragma unroll 2
    for (int k0 = 0; k0 < HH - 32; k0 += 32) {
        const int kn = k0 + 32;
        v16bf an = load_frag(Arow + kn, Arow + kn + 16);
        v16bf bn[4];
        #pragma unroll
        for (int g = 0; g < 4; ++g) bn[g] = load_frag(Br[g] + kn, Br[g] + kn + 16);
        #pragma unroll
        for (int g = 0; g < 4; ++g)
            acc[g] = __builtin_amdgcn_wmma_f32_16x16x32_bf16(
                false, a, false, bc[g], (short)0, acc[g], false, false);
        a = an;
        #pragma unroll
        for (int g = 0; g < 4; ++g) bc[g] = bn[g];
    }
    #pragma unroll
    for (int g = 0; g < 4; ++g)
        acc[g] = __builtin_amdgcn_wmma_f32_16x16x32_bf16(
            false, a, false, bc[g], (short)0, acc[g], false, false);

#if HAVE_ASYNC_LDS
#if __has_builtin(__builtin_amdgcn_s_wait_asynccnt)
    __builtin_amdgcn_s_wait_asynccnt(0);
#else
    asm volatile("s_wait_asynccnt 0" ::: "memory");
#endif
#endif
    __syncthreads();

    int col = cb + lr;
    #pragma unroll
    for (int i = 0; i < 8; ++i) {
        int row = m0 + i + ((lane >> 4) << 3);
        const float* xwr = &xw_tile[row * 64];
        float ai = acc[0][i] + xwr[0 * 16 + lr];
        float af = acc[1][i] + xwr[1 * 16 + lr];
        float ao = acc[2][i] + xwr[2 * 16 + lr];
        float ag = acc[3][i] + xwr[3 * 16 + lr];
        float ig = sigmoidf_(ai);
        float fg = sigmoidf_(af);
        float og = sigmoidf_(ao);
        float gg = tanhf(ag);
        size_t sidx = (size_t)row * HH + col;
        float cn = fg * c[sidx] + ig * gg;
        c[sidx] = cn;
        float hn = og * tanhf(cn);
        h_next[sidx] = (__bf16)hn;
        __builtin_nontemporal_store(hn, &out[((size_t)row * TT + t) * HH + col]);
    }
}

// ---------------------------------------------------------------------------
extern "C" void kernel_launch(void* const* d_in, const int* in_sizes, int n_in,
                              void* d_out, int out_size, void* d_ws, size_t ws_size,
                              hipStream_t stream) {
    const float* x  = (const float*)d_in[0];   // [N,T,D]
    const float* h0 = (const float*)d_in[1];   // [N,H]
    const float* Wx = (const float*)d_in[2];   // [D,4H]
    const float* Wh = (const float*)d_in[3];   // [H,4H]
    const float* b  = (const float*)d_in[4];   // [4H]
    float* out = (float*)d_out;                // [N,T,H]

    char* ws = (char*)d_ws;
    size_t off = 0;
    auto carve = [&](size_t bytes) -> void* {
        void* p = ws + off;
        off = (off + bytes + 255) & ~(size_t)255;
        return p;
    };
    __bf16* xb   = (__bf16*)carve((size_t)NB * TT * DD * sizeof(__bf16)); // 67 MB
    __bf16* WxT  = (__bf16*)carve((size_t)FH * DD * sizeof(__bf16));      // 8.4 MB
    __bf16* WhT  = (__bf16*)carve((size_t)FH * HH * sizeof(__bf16));      // 8.4 MB
    float*  xW   = (float*) carve((size_t)TT * NB * FH * sizeof(float));  // 537 MB
    __bf16* hb0  = (__bf16*)carve((size_t)NB * HH * sizeof(__bf16));
    __bf16* hb1  = (__bf16*)carve((size_t)NB * HH * sizeof(__bf16));
    float*  cbuf = (float*) carve((size_t)NB * HH * sizeof(float));
    (void)ws_size; (void)in_sizes; (void)n_in; (void)out_size;

    {
        int n = NB * TT * DD;
        cvt_f32_bf16<<<n / (256 * 4), 256, 0, stream>>>(x, xb, n);
    }
    transpose_cvt<<<dim3(FH / 32, DD / 32), 256, 0, stream>>>(Wx, WxT, DD, FH);
    transpose_cvt<<<dim3(FH / 32, HH / 32), 256, 0, stream>>>(Wh, WhT, HH, FH);
    {
        int n = NB * HH;
        cvt_f32_bf16<<<n / (256 * 4), 256, 0, stream>>>(h0, hb0, n);
        zero_f32<<<n / 256, 256, 0, stream>>>(cbuf, n);
    }
    gemm_xw<<<dim3((NB * TT) / 128, FH / 64), 128, 0, stream>>>(xb, WxT, b, xW);
    for (int t = 0; t < TT; ++t) {
        const __bf16* hp = (t & 1) ? hb1 : hb0;
        __bf16*       hn = (t & 1) ? hb0 : hb1;
        lstm_step<<<HH / 16, 128, 0, stream>>>(
            xW + (size_t)t * NB * FH, hp, WhT, cbuf, hn, out, t);
    }
}
